// SpatialCrossAttention_5377299054959
// MI455X (gfx1250) — matrix operand order repaired and checked by reading, not compiled, not run on hardware
//
#include <hip/hip_runtime.h>
#include <hip/hip_bf16.h>
#include <stdint.h>

// ---------------- problem constants ----------------
constexpr int B_  = 64;
constexpr int N_  = 312;
constexpr int D_  = 512;
constexpr int C_  = 1024;
constexpr int HW_ = 28 * 28;   // 784
constexpr int NH_ = 4;
constexpr int HD_ = 128;

typedef __attribute__((ext_vector_type(16))) __bf16 v16bf;
typedef __attribute__((ext_vector_type(8)))  __bf16 v8bf;
typedef __attribute__((ext_vector_type(8)))  float  v8f;
typedef __attribute__((ext_vector_type(4)))  unsigned int v4u;
typedef __attribute__((ext_vector_type(8)))  int v8i;
typedef __attribute__((ext_vector_type(4)))  int v4i;

__device__ __forceinline__ v8f vzero8() {
  v8f z = {0.f, 0.f, 0.f, 0.f, 0.f, 0.f, 0.f, 0.f};
  return z;
}

__device__ __forceinline__ __bf16 f2bf(float f) {
  unsigned u = __builtin_bit_cast(unsigned, f);
  u += 0x7FFFu + ((u >> 16) & 1u);          // round to nearest even
  unsigned short h = (unsigned short)(u >> 16);
  return __builtin_bit_cast(__bf16, h);
}

__device__ __forceinline__ __bf16 bf0() {
  return __builtin_bit_cast(__bf16, (unsigned short)0);
}

__device__ __forceinline__ v8bf vzero8bf() {
  v8bf z = {bf0(), bf0(), bf0(), bf0(), bf0(), bf0(), bf0(), bf0()};
  return z;
}

__device__ __forceinline__ v16bf cat8(v8bf lo, v8bf hi) {
  return __builtin_shufflevector(lo, hi, 0, 1, 2, 3, 4, 5, 6, 7,
                                 8, 9, 10, 11, 12, 13, 14, 15);
}

#define WMMA_BF16(a, b, c) \
  __builtin_amdgcn_wmma_f32_16x16x32_bf16(false, (a), false, (b), (short)0, (c), false, false)

// ---------------- TDM (Tensor Data Mover) support ----------------
#if defined(__AMDGCN__) && __has_builtin(__builtin_amdgcn_tensor_load_to_lds) && \
    __has_builtin(__builtin_amdgcn_s_wait_tensorcnt)
#define HAVE_TDM 1
#else
#define HAVE_TDM 0
#endif

#if HAVE_TDM
// DMA a 2D bf16 tile [tile_d1 rows x tile_d0 elems] from global (row stride
// `stride_elems`) into LDS at byte offset `lds_addr`, inserting 4 DWORDs of
// padding after each 16 DWORDs (64B row -> 80B LDS row stride).
// D# packing per CDNA5 ISA.
// 6-arg builtin (clang-23 form): (g0, g1, g2, g3, g4, cpol); unused
// descriptor groups zero-filled (2D tensor: groups 2/3 disabled).
__device__ __forceinline__ void tdm_load_2d_bf16_pad(
    unsigned lds_addr, const void* gptr, int tensor_d0, int tensor_d1,
    long long stride_elems, int tile_d0, int tile_d1) {
  unsigned long long ga = (unsigned long long)gptr;
  v4u g0;
  g0[0] = 1u;                                           // count=1 (user D#)
  g0[1] = lds_addr;                                     // [63:32] lds_addr
  g0[2] = (unsigned)(ga & 0xFFFFFFFFu);                 // [95:64]
  g0[3] = (unsigned)((ga >> 32) & 0x1FFFFFFu)           // [120:96]
          | (2u << 30);                                 // [127:126] type=2
  v8i g1;
  // data_size=1 (2B), pad_enable, pad_interval=3 (16 DW), pad_amount=3 (4 DW)
  g1[0] = (int)((1u << 16) | (1u << 20) | (3u << 22) | (3u << 25));
  g1[1] = (int)(((unsigned)tensor_d0 & 0xFFFFu) << 16); // dim0[15:0] @ [63:48]
  g1[2] = (int)(((unsigned)tensor_d0 >> 16) |
                (((unsigned)tensor_d1 & 0xFFFFu) << 16));
  g1[3] = (int)(((unsigned)tensor_d1 >> 16) |
                ((unsigned)tile_d0 << 16));             // tile_dim0 @ [127:112]
  g1[4] = (int)((unsigned)tile_d1 & 0xFFFFu);           // tile_dim1; tile_dim2=0
  g1[5] = (int)(unsigned)(stride_elems & 0xFFFFFFFFll); // dim0_stride[31:0]
  g1[6] = (int)(unsigned)((stride_elems >> 32) & 0xFFFFll);
  g1[7] = 0;
  v4i g2 = {0, 0, 0, 0};
  v4i g3 = {0, 0, 0, 0};
  v8i g4 = {0, 0, 0, 0, 0, 0, 0, 0};
  __builtin_amdgcn_tensor_load_to_lds(g0, g1, g2, g3, g4, 0);
}
#endif

// ============================================================
// Generic bf16 WMMA GEMM: C[M,N] = A[M,K] * B[K,N] + bias[N]
//   A: f32 or bf16, optionally transposed (A[m,k]=A[k*lda+m])
//   B: f32 weights, optionally transposed (B[k,n]=B[n*ldb+k])
//   C: bf16 out. Block tile 64x64, K-step 32, 4 waves (128 thr).
//   TDM_A (bf16, non-transposed, exact-multiple M/K only): the A
//   tile is DMA'd into LDS by the Tensor Data Mover (wave 0 issues,
//   TDM applies the 80B-row padding) while all waves stage B^T.
// ============================================================
template <bool A_F32, bool A_T, bool B_T, bool TDM_A>
__global__ void __launch_bounds__(128)
gemm_bf16(const void* __restrict__ Av, const float* __restrict__ Bw,
          const float* __restrict__ bias, __bf16* __restrict__ Cc,
          int M, int N, int K, int lda, int ldb, int ldc,
          long long strideA, long long strideC) {
  __shared__ __bf16 As[64][40];   // [m][k]  (80B row stride)
  __shared__ __bf16 Bst[64][40];  // [n][k]  (transposed stage)
  const int tid = threadIdx.x, wv = tid >> 5, lane = tid & 31;
  const int m0 = blockIdx.x * 64, n0 = blockIdx.y * 64;
  const long long zb = blockIdx.z;
  const float*  Af = (const float*)Av  + (A_F32 ? zb * strideA : 0);
  const __bf16* Ab = (const __bf16*)Av + (A_F32 ? 0 : zb * strideA);
  __bf16* Cp = Cc + zb * strideC;

  v8f acc[4] = {vzero8(), vzero8(), vzero8(), vzero8()};

  for (int k0 = 0; k0 < K; k0 += 32) {
    // ---- stage A tile [64][32] ----
    bool manualA = true;
#if HAVE_TDM
    if constexpr (TDM_A) {
      manualA = false;
      if (wv == 0) {
        tdm_load_2d_bf16_pad((unsigned)(uintptr_t)&As[0][0],
                             Ab + (long long)m0 * lda + k0,
                             K - k0, M - m0, (long long)lda, 32, 64);
      }
    }
#endif
    if (manualA) {
#pragma unroll
      for (int t2 = 0; t2 < 2; ++t2) {
        int chunk = tid + t2 * 128;   // 0..255
        int r = chunk >> 2;           // 0..63
        int c8 = (chunk & 3) * 8;     // 0,8,16,24
        int m = m0 + r;
        v8bf val;
#pragma unroll
        for (int j = 0; j < 8; ++j) {
          int k = k0 + c8 + j;
          bool ok = (m < M) && (k < K);
          if (A_F32) {
            float f = ok ? (A_T ? Af[(long long)k * lda + m]
                                : Af[(long long)m * lda + k])
                         : 0.f;
            val[j] = f2bf(f);
          } else {
            val[j] = ok ? (A_T ? Ab[(long long)k * lda + m]
                               : Ab[(long long)m * lda + k])
                        : bf0();
          }
        }
        *(v8bf*)&As[r][c8] = val;
      }
    }
    // ---- stage B tile transposed (overlaps the TDM copy) ----
#pragma unroll
    for (int t2 = 0; t2 < 2; ++t2) {
      int chunk = tid + t2 * 128;   // 0..255
      int kr = chunk >> 3;          // 0..31
      int nc = (chunk & 7) * 8;     // 0..56
      int k = k0 + kr;
#pragma unroll
      for (int j = 0; j < 8; ++j) {
        int n = n0 + nc + j;
        float f = 0.f;
        if (k < K && n < N)
          f = B_T ? Bw[(long long)n * ldb + k] : Bw[(long long)k * ldb + n];
        Bst[nc + j][kr] = f2bf(f);
      }
    }
#if HAVE_TDM
    if constexpr (TDM_A) __builtin_amdgcn_s_wait_tensorcnt(0);
#endif
    __syncthreads();

    // ---- fragments: contiguous 16B LDS loads ----
    const int mr = wv * 16 + (lane & 15);
    const int kb = (lane >> 4) << 4;   // 0 or 16
    v16bf af = cat8(*(const v8bf*)&As[mr][kb], *(const v8bf*)&As[mr][8 + kb]);
    const int kb2 = (lane >> 4) << 4;  // 0 or 16
#pragma unroll
    for (int cf = 0; cf < 4; ++cf) {
      int nn = cf * 16 + (lane & 15);
      v16bf bfr = cat8(*(const v8bf*)&Bst[nn][kb2],
                       *(const v8bf*)&Bst[nn][kb2 + 8]);
      acc[cf] = WMMA_BF16(af, bfr, acc[cf]);
    }
    __syncthreads();
  }

  // ---- store C (bf16, +bias) ----
#pragma unroll
  for (int cf = 0; cf < 4; ++cf) {
    int n = n0 + cf * 16 + (lane & 15);
#pragma unroll
    for (int j = 0; j < 8; ++j) {
      int m = m0 + wv * 16 + j + ((lane >> 4) << 3);
      if (m < M && n < N) {
        float r = acc[cf][j] + (bias ? bias[n] : 0.f);
        Cp[(long long)m * ldc + n] = f2bf(r);
      }
    }
  }
}

// ============================================================
// Fused attention: one block per (n-tile of 16, head, batch).
// Full [16 x 784] logits kept in LDS (f32, row padded to 808),
// softmax in LDS, head-mean atomically accumulated into
// concept_maps, then attn @ V via WMMA (V staged transposed).
// 128 threads = 4 waves.
// ============================================================
__global__ void __launch_bounds__(128)
attn_kernel(const __bf16* __restrict__ q, const __bf16* __restrict__ kv,
            float* __restrict__ att_out, float* __restrict__ cmap) {
  constexpr int LGW = 808;              // padded logits row
  __shared__ float  lg[16][LGW];        // logits -> attn probs
  __shared__ __bf16 vst[128][40];       // V slab transposed: [d][s]
  __shared__ float  red[16][8];
  __shared__ float  rmax_s[16];
  __shared__ float  rinv_s[16];

  const int tid = threadIdx.x, wv = tid >> 5, lane = tid & 31;
  const int n0 = blockIdx.x * 16, h = blockIdx.y, b = blockIdx.z;
  const float scale = 0.08838834764831845f;  // 1/sqrt(128)

  // ---------- phase 1: logits = q @ k^T ----------
  v16bf aq[4];
  {
    int mrow = n0 + (lane & 15);
    bool valid = mrow < N_;
    const __bf16* qrow =
        q + (long long)(b * N_ + (valid ? mrow : 0)) * D_ + h * HD_;
    int kb = (lane >> 4) << 4;
#pragma unroll
    for (int kc = 0; kc < 4; ++kc) {
      int kk0 = kc * 32 + kb;
      v8bf lo = valid ? *(const v8bf*)(qrow + kk0) : vzero8bf();
      v8bf hi = valid ? *(const v8bf*)(qrow + kk0 + 8) : vzero8bf();
      aq[kc] = cat8(lo, hi);
    }
  }
  for (int st = wv; st < 49; st += 4) {
    int s0 = st * 16;
    v8f acc = vzero8();
    int scol = s0 + (lane & 15);
    const __bf16* kp =
        kv + (long long)(b * HW_ + scol) * (2 * D_) + h * HD_ + ((lane >> 4) << 4);
#pragma unroll
    for (int kc = 0; kc < 4; ++kc) {
      v16bf bk = cat8(*(const v8bf*)(kp + kc * 32),
                      *(const v8bf*)(kp + kc * 32 + 8));
      acc = WMMA_BF16(aq[kc], bk, acc);
    }
#pragma unroll
    for (int j = 0; j < 8; ++j)
      lg[j + ((lane >> 4) << 3)][s0 + (lane & 15)] = acc[j] * scale;
  }
  __syncthreads();

  // ---------- phase 2: softmax over 784 cols, emit concept maps ----------
  {
    int row = tid >> 3, sub = tid & 7;  // 16 rows x 8 threads
    float mx = -3.4e38f;
    for (int c = sub; c < HW_; c += 8) mx = fmaxf(mx, lg[row][c]);
    red[row][sub] = mx;
    __syncthreads();
    if (sub == 0) {
      float m2 = red[row][0];
#pragma unroll
      for (int i = 1; i < 8; ++i) m2 = fmaxf(m2, red[row][i]);
      rmax_s[row] = m2;
    }
    __syncthreads();
    float rm = rmax_s[row], s = 0.f;
    for (int c = sub; c < HW_; c += 8) {
      float e = __expf(lg[row][c] - rm);
      lg[row][c] = e;
      s += e;
    }
    red[row][sub] = s;
    __syncthreads();
    if (sub == 0) {
      float t = 0.f;
#pragma unroll
      for (int i = 0; i < 8; ++i) t += red[row][i];
      rinv_s[row] = 1.f / t;
    }
    __syncthreads();
    float inv = rinv_s[row];
    for (int c = sub; c < HW_; c += 8) lg[row][c] *= inv;
    for (int c = HW_ + sub; c < 800; c += 8) lg[row][c] = 0.f;
    if (n0 + row < N_) {
      float* cm = cmap + (long long)(b * N_ + n0 + row) * HW_;
      for (int c = sub; c < HW_; c += 8) atomicAdd(&cm[c], lg[row][c] * 0.25f);
    }
  }
  __syncthreads();

  // ---------- phase 3: attended = attn @ v ----------
  v8f av0 = vzero8(), av1 = vzero8();
  for (int sc = 0; sc < 25; ++sc) {
    // stage V transposed: read rows of kv (contiguous d), scatter into [d][s]
#pragma unroll
    for (int t2 = 0; t2 < 4; ++t2) {
      int chunk = tid + t2 * 128;    // 0..511
      int r = chunk >> 4;            // s in 0..31
      int c8 = (chunk & 15) * 8;     // d in 0..120
      int s = sc * 32 + r;
      v8bf tv = vzero8bf();
      if (s < HW_)
        tv = *(const v8bf*)(kv + (long long)(b * HW_ + s) * (2 * D_) + D_ +
                            h * HD_ + c8);
#pragma unroll
      for (int j = 0; j < 8; ++j) vst[c8 + j][r] = tv[j];
    }
    __syncthreads();
    // A fragment from attn probs (f32 -> bf16)
    v16bf aa;
    {
      int kb = (lane >> 4) << 4, mm = lane & 15;
      const float* lrow = &lg[mm][sc * 32];
      float4 f0 = *(const float4*)(lrow + kb);
      float4 f1 = *(const float4*)(lrow + kb + 4);
      float4 f2 = *(const float4*)(lrow + kb + 8);
      float4 f3 = *(const float4*)(lrow + kb + 12);
      aa[0] = f2bf(f0.x);  aa[1] = f2bf(f0.y);
      aa[2] = f2bf(f0.z);  aa[3] = f2bf(f0.w);
      aa[4] = f2bf(f1.x);  aa[5] = f2bf(f1.y);
      aa[6] = f2bf(f1.z);  aa[7] = f2bf(f1.w);
      aa[8] = f2bf(f2.x);  aa[9] = f2bf(f2.y);
      aa[10] = f2bf(f2.z); aa[11] = f2bf(f2.w);
      aa[12] = f2bf(f3.x); aa[13] = f2bf(f3.y);
      aa[14] = f2bf(f3.z); aa[15] = f2bf(f3.w);
    }
    const int kb2 = (lane >> 4) << 4;
    {
      int nn = wv * 32 + (lane & 15);
      v16bf bv = cat8(*(const v8bf*)&vst[nn][kb2],
                      *(const v8bf*)&vst[nn][kb2 + 8]);
      av0 = WMMA_BF16(aa, bv, av0);
    }
    {
      int nn = wv * 32 + 16 + (lane & 15);
      v16bf bv = cat8(*(const v8bf*)&vst[nn][kb2],
                      *(const v8bf*)&vst[nn][kb2 + 8]);
      av1 = WMMA_BF16(aa, bv, av1);
    }
    __syncthreads();
  }
  // store attended (f32)
#pragma unroll
  for (int j = 0; j < 8; ++j) {
    int mrow = n0 + j + ((lane >> 4) << 3);
    if (mrow < N_) {
      long long base = (long long)(b * N_ + mrow) * D_ + h * HD_;
      att_out[base + wv * 32 + (lane & 15)]      = av0[j];
      att_out[base + wv * 32 + 16 + (lane & 15)] = av1[j];
    }
  }
}

// ============================================================
// Fused residual + LayerNorm + LayerNorm2  ->  z (bf16)
// one block (128 thr) per row of [B*N, 512]
// ============================================================
__device__ __forceinline__ float block_sum(float v, float* red) {
  int tid = threadIdx.x;
  red[tid] = v;
  __syncthreads();
  for (int off = 64; off > 0; off >>= 1) {
    if (tid < off) red[tid] += red[tid + off];
    __syncthreads();
  }
  float r = red[0];
  __syncthreads();
  return r;
}

__global__ void __launch_bounds__(128)
ln_kernel(const float* __restrict__ att, const float* __restrict__ ce,
          const float* __restrict__ g1, const float* __restrict__ bb1,
          const float* __restrict__ g2, const float* __restrict__ bb2,
          __bf16* __restrict__ z) {
  __shared__ float red[128];
  const long long row = blockIdx.x;
  const int tid = threadIdx.x;
  const float* xa = att + row * D_;
  const float* xc = ce + row * D_;

  float x[4];
#pragma unroll
  for (int i = 0; i < 4; ++i) {
    int c = tid + i * 128;
    x[i] = xa[c] + xc[c];
  }
  float s = x[0] + x[1] + x[2] + x[3];
  float ss = x[0] * x[0] + x[1] * x[1] + x[2] * x[2] + x[3] * x[3];
  float mean = block_sum(s, red) * (1.f / D_);
  float var = block_sum(ss, red) * (1.f / D_) - mean * mean;
  float r1 = rsqrtf(var + 1e-5f);

  float y[4];
  float s2 = 0.f, ss2 = 0.f;
#pragma unroll
  for (int i = 0; i < 4; ++i) {
    int c = tid + i * 128;
    y[i] = (x[i] - mean) * r1 * g1[c] + bb1[c];
    s2 += y[i];
    ss2 += y[i] * y[i];
  }
  float mean2 = block_sum(s2, red) * (1.f / D_);
  float var2 = block_sum(ss2, red) * (1.f / D_) - mean2 * mean2;
  float r2 = rsqrtf(var2 + 1e-5f);
#pragma unroll
  for (int i = 0; i < 4; ++i) {
    int c = tid + i * 128;
    z[row * D_ + c] = f2bf((y[i] - mean2) * r2 * g2[c] + bb2[c]);
  }
}

// ============================================================
// One-off: transpose W1 (f32 [512,1024]) -> bf16 [1024][512]
// so score-kernel B fragments are contiguous 16B loads.
// ============================================================
__global__ void __launch_bounds__(256)
w1_transpose(const float* __restrict__ W1, __bf16* __restrict__ w1t) {
  int idx = blockIdx.x * 256 + threadIdx.x;   // over 512*1024
  int k = idx >> 10;        // 0..511
  int n = idx & 1023;       // 0..1023
  w1t[(long long)n * D_ + k] = f2bf(W1[idx]);
}

// ============================================================
// Fused score MLP: sigmoid( gelu(z @ W1 + b1) @ W2 + b2 )
// one block (128 thr = 4 waves) per 16 rows; hidden never stored.
// ============================================================
__global__ void __launch_bounds__(128)
score_kernel(const __bf16* __restrict__ z, const __bf16* __restrict__ w1t,
             const float* __restrict__ bb1, const float* __restrict__ W2,
             const float* __restrict__ bb2, float* __restrict__ out) {
  __shared__ __bf16 zb[16][520];   // padded: 1040B row stride
  __shared__ float rowsum[16];
  const int tid = threadIdx.x, wv = tid >> 5, lane = tid & 31;
  const long long r0 = (long long)blockIdx.x * 16;

  for (int i = tid; i < 16 * 512; i += 128) {
    int r = i >> 9, c = i & 511;
    zb[r][c] = z[(r0 + r) * D_ + c];
  }
  if (tid < 16) rowsum[tid] = 0.f;
  __syncthreads();

  float part[8] = {0.f, 0.f, 0.f, 0.f, 0.f, 0.f, 0.f, 0.f};
  const int kb = (lane >> 4) << 3;    // 0 or 8
  const int kb2 = (lane >> 4) << 4;   // 0 or 16
  const int mm = lane & 15;
  for (int nt = wv; nt < 64; nt += 4) {
    v8f acc = vzero8();
    int col = nt * 16 + (lane & 15);
    const __bf16* wrow = w1t + (long long)col * D_ + kb2;
    for (int kc = 0; kc < 16; ++kc) {
      int kk0 = kc * 32 + kb2;
      v16bf af = cat8(*(const v8bf*)&zb[mm][kk0],
                      *(const v8bf*)&zb[mm][kk0 + 8]);
      v16bf bfr = cat8(*(const v8bf*)(wrow + kc * 32),
                       *(const v8bf*)(wrow + kc * 32 + 8));
      acc = WMMA_BF16(af, bfr, acc);
    }
    float w2v = W2[col], b1v = bb1[col];
#pragma unroll
    for (int j = 0; j < 8; ++j) {
      float xx = acc[j] + b1v;
      float gel = 0.5f * xx * (1.f + erff(xx * 0.7071067811865476f));
      part[j] += gel * w2v;
    }
  }
#pragma unroll
  for (int j = 0; j < 8; ++j)
    atomicAdd(&rowsum[j + ((lane >> 4) << 3)], part[j]);
  __syncthreads();
  if (tid < 16) {
    float sc = rowsum[tid] + bb2[0];
    out[r0 + tid] = 1.f / (1.f + __expf(-sc));
  }
}

// ============================================================
extern "C" void kernel_launch(void* const* d_in, const int* in_sizes, int n_in,
                              void* d_out, int out_size, void* d_ws, size_t ws_size,
                              hipStream_t stream) {
  (void)in_sizes; (void)n_in; (void)out_size; (void)ws_size;
  const float* features = (const float*)d_in[0];
  const float* c_emb    = (const float*)d_in[1];
  const float* W_fp = (const float*)d_in[2];
  const float* b_fp = (const float*)d_in[3];
  const float* W_q  = (const float*)d_in[4];
  const float* b_q  = (const float*)d_in[5];
  const float* W_mc = (const float*)d_in[6];
  const float* b_mc = (const float*)d_in[7];
  const float* W_kv = (const float*)d_in[8];
  const float* b_kv = (const float*)d_in[9];
  const float* ln_g  = (const float*)d_in[10];
  const float* ln_b  = (const float*)d_in[11];
  const float* ln2_g = (const float*)d_in[12];
  const float* ln2_b = (const float*)d_in[13];
  const float* W1 = (const float*)d_in[14];
  const float* b1 = (const float*)d_in[15];
  const float* W2 = (const float*)d_in[16];
  const float* b2 = (const float*)d_in[17];

  float* out_scores = (float*)d_out;
  float* out_maps   = out_scores + (size_t)B_ * N_;

  // workspace carve-up (~257 MB)
  char* ws = (char*)d_ws;
  size_t off = 0;
  __bf16* fp  = (__bf16*)(ws + off); off += (size_t)B_ * HW_ * D_ * 2;      // 51.4 MB
  __bf16* kvb = (__bf16*)(ws + off); off += (size_t)B_ * HW_ * 2 * D_ * 2;  // 102.8 MB
  __bf16* q1  = (__bf16*)(ws + off); off += (size_t)B_ * N_ * D_ * 2;       // 20.4 MB
  __bf16* q2  = (__bf16*)(ws + off); off += (size_t)B_ * N_ * D_ * 2;       // 20.4 MB
  float*  att = (float*)(ws + off);  off += (size_t)B_ * N_ * D_ * 4;       // 40.9 MB
  __bf16* zbf = (__bf16*)(ws + off); off += (size_t)B_ * N_ * D_ * 2;       // 20.4 MB
  __bf16* w1t = (__bf16*)(ws + off); off += (size_t)2 * D_ * D_ * 2;        // 1.0 MB

  // concept_maps accumulated with atomics -> zero first (capturable memset)
  (void)hipMemsetAsync(out_maps, 0, (size_t)B_ * N_ * HW_ * sizeof(float), stream);

  dim3 blk(128);
  // one-off weight transpose for score MLP
  w1_transpose<<<dim3((2 * D_ * D_) / 256), dim3(256), 0, stream>>>(W1, w1t);
  // fp[b,hw,d] = features[b,:,hw] . W_fp[d,:] + b_fp   (A trans f32, B trans)
  gemm_bf16<true, true, true, false><<<dim3(13, 8, 64), blk, 0, stream>>>(
      features, W_fp, b_fp, fp, HW_, D_, C_, HW_, C_, D_,
      (long long)C_ * HW_, (long long)HW_ * D_);
  // kv = fp @ W_kv + b_kv   [B*HW, 1024]   (A staged by TDM)
  gemm_bf16<false, false, false, true><<<dim3(784, 16, 1), blk, 0, stream>>>(
      fp, W_kv, b_kv, kvb, B_ * HW_, 2 * D_, D_, D_, 2 * D_, 2 * D_, 0, 0);
  // q1 = c_emb @ W_q + b_q
  gemm_bf16<true, false, false, false><<<dim3(312, 8, 1), blk, 0, stream>>>(
      c_emb, W_q, b_q, q1, B_ * N_, D_, D_, D_, D_, D_, 0, 0);
  // q2 = q1 @ W_mc + b_mc   (A staged by TDM)
  gemm_bf16<false, false, false, true><<<dim3(312, 8, 1), blk, 0, stream>>>(
      q1, W_mc, b_mc, q2, B_ * N_, D_, D_, D_, D_, D_, 0, 0);
  // fused attention (+softmax, +concept_maps, +attn@V)
  attn_kernel<<<dim3(20, NH_, B_), blk, 0, stream>>>(q2, kvb, att, out_maps);
  // residual + LN + LN2 -> z
  ln_kernel<<<dim3(B_ * N_), blk, 0, stream>>>(att, c_emb, ln_g, ln_b, ln2_g,
                                               ln2_b, zbf);
  // fused score MLP -> scores
  score_kernel<<<dim3((B_ * N_) / 16), blk, 0, stream>>>(zbf, w1t, b1, W2, b2,
                                                         out_scores);
}
